// GNN_MLP_model_80642305949836
// MI455X (gfx1250) — matrix operand order repaired and checked by reading, not compiled, not run on hardware
//
#include <hip/hip_runtime.h>
#include <hip/hip_bf16.h>
#include <math.h>

// ---------------------------------------------------------------------------
// Types for CDNA5 WMMA (wave32, v_wmma_f32_16x16x32_bf16)
// ---------------------------------------------------------------------------
typedef __attribute__((ext_vector_type(16))) __bf16 v16bf;
typedef __attribute__((ext_vector_type(8)))  __bf16 v8bf;
typedef __attribute__((ext_vector_type(4)))  __bf16 v4bf;
typedef __attribute__((ext_vector_type(8)))  float  v8f;

static __device__ __forceinline__ __bf16 f2bf(float f) {
  unsigned u = __builtin_bit_cast(unsigned, f);
  u += 0x7FFFu + ((u >> 16) & 1u);              // round-to-nearest-even
  unsigned short h = (unsigned short)(u >> 16);
  return __builtin_bit_cast(__bf16, h);
}
static __device__ __forceinline__ float bf2f(__bf16 b) {
  unsigned short h = __builtin_bit_cast(unsigned short, b);
  unsigned u = ((unsigned)h) << 16;
  return __builtin_bit_cast(float, u);
}

static __device__ __forceinline__ v8f wmma_bf16(v16bf a, v16bf b, v8f c) {
  // (neg_a, A, neg_b, B, c_mod, C, reuse_a, reuse_b)
  return __builtin_amdgcn_wmma_f32_16x16x32_bf16(false, a, false, b, (short)0, c,
                                                 false, false);
}

// A fragment: 16x32 bf16, row-major source. Lane L (g=L/16, m=L%16) holds
// row m, K = k0+8g+{0..7} (elems 0..7) and k0+16+8g+{0..7} (elems 8..15).
static __device__ __forceinline__ v16bf load_a(const __bf16* rowp, int k0, int g) {
  v8bf lo = *(const v8bf*)(rowp + k0 + 8 * g);
  v8bf hi = *(const v8bf*)(rowp + k0 + 16 + 8 * g);
  return __builtin_shufflevector(lo, hi, 0,1,2,3,4,5,6,7,8,9,10,11,12,13,14,15);
}
// B fragment (two 8-bf16 chunks at p and p+16 per lane), source may be LDS or global.
static __device__ __forceinline__ v16bf load_b(const __bf16* p) {
  v8bf lo = *(const v8bf*)p;
  v8bf hi = *(const v8bf*)(p + 16);
  return __builtin_shufflevector(lo, hi, 0,1,2,3,4,5,6,7,8,9,10,11,12,13,14,15);
}

// Padded LDS block stride: 32 payload bf16 + 8 pad = 80B. Keeps 16B alignment
// for per-lane b128 reads and spreads the 16 column-lanes across banks.
#define BSTRIDE 40

// ---------------------------------------------------------------------------
// Elementwise / prep kernels
// ---------------------------------------------------------------------------
__global__ void pack_bf16_kernel(const float* __restrict__ x,
                                 __bf16* __restrict__ y, long n) {
  long i = (long)blockIdx.x * blockDim.x + threadIdx.x;
  if (i < n) y[i] = f2bf(x[i]);
}

// Pack weight [K x C] row-major -> B-fragment panels: out[(kt*C + c)*32 + kk]
__global__ void pack_w_kernel(const float* __restrict__ W,
                              __bf16* __restrict__ out, int K, int C) {
  long i = (long)blockIdx.x * blockDim.x + threadIdx.x;
  if (i >= (long)K * C) return;
  int k = (int)(i / C), c = (int)(i % C);
  int kt = k >> 5, kk = k & 31;
  out[((long)kt * C + c) * 32 + kk] = f2bf(W[(long)k * C + c]);
}

__global__ void deg_count_kernel(const int* __restrict__ dst,
                                 float* __restrict__ deg, int E) {
  int i = blockIdx.x * blockDim.x + threadIdx.x;
  if (i < E) atomicAdd(&deg[dst[i]], 1.0f);
}

__global__ void inv_deg_kernel(float* __restrict__ deg, int N) {
  int i = blockIdx.x * blockDim.x + threadIdx.x;
  if (i < N) deg[i] = 1.0f / fmaxf(deg[i], 1.0f);
}

// agg[dst] += x[src], f32 source, 4 channels per thread
__global__ void scatter_f32_kernel(const float* __restrict__ X,
                                   const int* __restrict__ src,
                                   const int* __restrict__ dst,
                                   float* __restrict__ agg, int E, int C) {
  long tid = (long)blockIdx.x * blockDim.x + threadIdx.x;
  int per = C >> 2;
  long e = tid / per;
  if (e >= E) return;
  int c = (int)(tid % per) << 2;
  int s = src[e], d = dst[e];
  const float4 v = *(const float4*)(X + (long)s * C + c);
  float* p = agg + (long)d * C + c;
  atomicAdd(p + 0, v.x);
  atomicAdd(p + 1, v.y);
  atomicAdd(p + 2, v.z);
  atomicAdd(p + 3, v.w);
}

// agg[dst] += x[src], bf16 source
__global__ void scatter_bf16_kernel(const __bf16* __restrict__ X,
                                    const int* __restrict__ src,
                                    const int* __restrict__ dst,
                                    float* __restrict__ agg, int E, int C) {
  long tid = (long)blockIdx.x * blockDim.x + threadIdx.x;
  int per = C >> 2;
  long e = tid / per;
  if (e >= E) return;
  int c = (int)(tid % per) << 2;
  int s = src[e], d = dst[e];
  const v4bf v = *(const v4bf*)(X + (long)s * C + c);
  float* p = agg + (long)d * C + c;
  atomicAdd(p + 0, bf2f(v[0]));
  atomicAdd(p + 1, bf2f(v[1]));
  atomicAdd(p + 2, bf2f(v[2]));
  atomicAdd(p + 3, bf2f(v[3]));
}

// y = bf16(agg * inv_deg[row])
__global__ void scale_pack_kernel(const float* __restrict__ agg,
                                  const float* __restrict__ inv,
                                  __bf16* __restrict__ y, int N, int C) {
  long i = (long)blockIdx.x * blockDim.x + threadIdx.x;
  if (i >= (long)N * C) return;
  int r = (int)(i / C);
  y[i] = f2bf(agg[i] * inv[r]);
}

// ---------------------------------------------------------------------------
// Fused SAGE layer: out = relu(Xs @ Ws + Xn @ Wn + bias), K=128 fixed.
// Persistent blocks: stage both packed weight matrices + bias into LDS once,
// then grid-stride over 16-row tiles; one wave per tile, all B from LDS.
// ---------------------------------------------------------------------------
template <int NT, bool WRITE_F32>
__global__ __launch_bounds__(256)
void sage_gemm_kernel(const __bf16* __restrict__ Xs, const __bf16* __restrict__ Xn,
                      const __bf16* __restrict__ Ws, const __bf16* __restrict__ Wn,
                      const float* __restrict__ bias,
                      __bf16* __restrict__ out_bf, float* __restrict__ out_f,
                      int nrows) {
  constexpr int K = 128;
  constexpr int NC = NT * 16;
  constexpr int NBLK = (K / 32) * NC;   // 32-deep column panels per matrix
  __shared__ __align__(16) __bf16 Wlds[2 * NBLK * BSTRIDE];
  __shared__ float blds[NC];

  for (int i = threadIdx.x; i < NBLK * 32; i += 256) {
    int blk = i >> 5, kk = i & 31;
    Wlds[blk * BSTRIDE + kk] = Ws[i];
    Wlds[(NBLK + blk) * BSTRIDE + kk] = Wn[i];
  }
  for (int i = threadIdx.x; i < NC; i += 256) blds[i] = bias[i];
  __syncthreads();

  int lane = threadIdx.x & 31;
  int wv = threadIdx.x >> 5;           // 8 waves per block
  int g = lane >> 4, n = lane & 15;
  int ntiles = (nrows + 15) >> 4;

  for (int mt = blockIdx.x * 8 + wv; mt < ntiles; mt += (int)gridDim.x * 8) {
    int m0 = mt * 16;
    int arow = m0 + n;
    if (arow >= nrows) arow = nrows - 1;

    v8f acc[NT];
#pragma unroll
    for (int t = 0; t < NT; ++t) acc[t] = (v8f){0, 0, 0, 0, 0, 0, 0, 0};

    const __bf16* xr = Xs + (long)arow * K;
#pragma unroll
    for (int kt = 0; kt < K / 32; ++kt) {
      v16bf a = load_a(xr, kt * 32, g);
#pragma unroll
      for (int t = 0; t < NT; ++t) {
        v16bf b = load_b(&Wlds[(kt * NC + t * 16 + n) * BSTRIDE + 8 * g]);
        acc[t] = wmma_bf16(a, b, acc[t]);
      }
    }
    xr = Xn + (long)arow * K;
#pragma unroll
    for (int kt = 0; kt < K / 32; ++kt) {
      v16bf a = load_a(xr, kt * 32, g);
#pragma unroll
      for (int t = 0; t < NT; ++t) {
        v16bf b = load_b(&Wlds[(NBLK + kt * NC + t * 16 + n) * BSTRIDE + 8 * g]);
        acc[t] = wmma_bf16(a, b, acc[t]);
      }
    }

    // C/D layout: lane(g,n) elem v = row m0+8g+v, col n (+16t)
#pragma unroll
    for (int t = 0; t < NT; ++t) {
      int col = t * 16 + n;
      float bc = blds[col];
#pragma unroll
      for (int v = 0; v < 8; ++v) {
        int r = m0 + 8 * g + v;
        if (r < nrows) {
          float val = acc[t][v] + bc;
          val = val > 0.0f ? val : 0.0f;
          out_bf[(long)r * NC + col] = f2bf(val);
          if (WRITE_F32) out_f[(long)r * NC + col] = val;
        }
      }
    }
  }
}

// ---------------------------------------------------------------------------
// Edge scorer: persistent blocks stage W_mlp1 panels (+b1,+w2) into LDS once,
// then grid-stride over 16-edge tiles. Per tile: gather A = concat(h[src],
// h[dst]) (16x128 bf16), 32 WMMAs for the hidden layer, then 128->1 dot +
// sigmoid via cross-lane shfl_xor reduction (wave32).
// ---------------------------------------------------------------------------
__global__ __launch_bounds__(256)
void edge_score_kernel(const __bf16* __restrict__ H,  // [N, 64]
                       const int* __restrict__ esrc, const int* __restrict__ edst,
                       const __bf16* __restrict__ Wp,  // packed [4][128][32]
                       const float* __restrict__ b1v, const float* __restrict__ w2,
                       const float* __restrict__ b2, float* __restrict__ out,
                       int E) {
  constexpr int NBLK = 4 * 128;
  __shared__ __align__(16) __bf16 Wlds[NBLK * BSTRIDE];
  __shared__ float blds[128], wlds[128];

  for (int i = threadIdx.x; i < NBLK * 32; i += 256) {
    int blk = i >> 5, kk = i & 31;
    Wlds[blk * BSTRIDE + kk] = Wp[i];
  }
  for (int i = threadIdx.x; i < 128; i += 256) {
    blds[i] = b1v[i];
    wlds[i] = w2[i];
  }
  __syncthreads();

  int lane = threadIdx.x & 31;
  int wv = threadIdx.x >> 5;
  int g = lane >> 4, n = lane & 15;
  int ntiles = (E + 15) >> 4;
  float bb = b2[0];

  for (int et = blockIdx.x * 8 + wv; et < ntiles; et += (int)gridDim.x * 8) {
    int e0 = et * 16;
    int em = e0 + n;
    if (em >= E) em = E - 1;
    long sbase = (long)esrc[em] * 64;
    long dbase = (long)edst[em] * 64;

    v8f acc[8];
#pragma unroll
    for (int t = 0; t < 8; ++t) acc[t] = (v8f){0, 0, 0, 0, 0, 0, 0, 0};

#pragma unroll
    for (int kt = 0; kt < 4; ++kt) {
      int ka = kt * 32 + 8 * g;        // 8-elem chunk: stays within one half
      int kb = ka + 16;
      const __bf16* pa = (ka < 64) ? (H + sbase + ka) : (H + dbase + (ka - 64));
      const __bf16* pb = (kb < 64) ? (H + sbase + kb) : (H + dbase + (kb - 64));
      v8bf lo = *(const v8bf*)pa;
      v8bf hi = *(const v8bf*)pb;
      v16bf a = __builtin_shufflevector(lo, hi, 0,1,2,3,4,5,6,7,8,9,10,11,12,13,14,15);
#pragma unroll
      for (int t = 0; t < 8; ++t) {
        v16bf b = load_b(&Wlds[(kt * 128 + t * 16 + n) * BSTRIDE + 8 * g]);
        acc[t] = wmma_bf16(a, b, acc[t]);
      }
    }

    float partial[8];
#pragma unroll
    for (int v = 0; v < 8; ++v) partial[v] = 0.0f;
#pragma unroll
    for (int t = 0; t < 8; ++t) {
      int j = t * 16 + n;
      float bj = blds[j], wj = wlds[j];
#pragma unroll
      for (int v = 0; v < 8; ++v) {
        float r = acc[t][v] + bj;
        r = r > 0.0f ? r : 0.0f;
        partial[v] += r * wj;
      }
    }
    // reduce over the 16 lanes of each half-wave (columns j)
#pragma unroll
    for (int off = 1; off < 16; off <<= 1) {
#pragma unroll
      for (int v = 0; v < 8; ++v)
        partial[v] += __shfl_xor(partial[v], off, 32);
    }
    if (n == 0) {
#pragma unroll
      for (int v = 0; v < 8; ++v) {
        int e = e0 + 8 * g + v;
        if (e < E) out[e] = 1.0f / (1.0f + expf(-(partial[v] + bb)));
      }
    }
  }
}

// ---------------------------------------------------------------------------
// Host launch
// ---------------------------------------------------------------------------
static inline long cdiv(long a, long b) { return (a + b - 1) / b; }

extern "C" void kernel_launch(void* const* d_in, const int* in_sizes, int n_in,
                              void* d_out, int out_size, void* d_ws, size_t ws_size,
                              hipStream_t stream) {
  const int IN = 128, H1 = 128, H2 = 64, MH = 128;
  const int N  = in_sizes[0] / IN;
  const int E1 = in_sizes[1];
  const int E2 = in_sizes[3];

  const float* inputs   = (const float*)d_in[0];
  const int* edge_src   = (const int*)d_in[1];
  const int* edge_dst   = (const int*)d_in[2];
  const int* score_src  = (const int*)d_in[3];
  const int* score_dst  = (const int*)d_in[4];
  const float* W_self1  = (const float*)d_in[5];
  const float* W_neigh1 = (const float*)d_in[6];
  const float* b1       = (const float*)d_in[7];
  const float* W_self2  = (const float*)d_in[8];
  const float* W_neigh2 = (const float*)d_in[9];
  const float* b2       = (const float*)d_in[10];
  const float* W_mlp1   = (const float*)d_in[11];
  const float* b_mlp1   = (const float*)d_in[12];
  const float* W_mlp2   = (const float*)d_in[13];
  const float* b_mlp2   = (const float*)d_in[14];

  float* out_score = (float*)d_out;            // [E2]
  float* out_h     = (float*)d_out + E2;       // [N, 64]

  // Workspace carve-up (256B aligned slabs)
  char* ws = (char*)d_ws;
  size_t off = 0;
  auto carve = [&](size_t bytes) -> char* {
    char* p = ws + off;
    off += (bytes + 255) & ~(size_t)255;
    return p;
  };
  float*  deg    = (float*)carve((size_t)N * 4);              // deg -> inv_deg in place
  float*  aggf   = (float*)carve((size_t)N * IN * 4);         // f32 accumulators (reused)
  __bf16* Xbf    = (__bf16*)carve((size_t)N * IN * 2);
  __bf16* AGGbf  = (__bf16*)carve((size_t)N * IN * 2);        // reused for both layers
  __bf16* H1bf   = (__bf16*)carve((size_t)N * H1 * 2);
  __bf16* Hbf    = (__bf16*)carve((size_t)N * H2 * 2);
  __bf16* Ws1p   = (__bf16*)carve((size_t)IN * H1 * 2);
  __bf16* Wn1p   = (__bf16*)carve((size_t)IN * H1 * 2);
  __bf16* Ws2p   = (__bf16*)carve((size_t)H1 * H2 * 2);
  __bf16* Wn2p   = (__bf16*)carve((size_t)H1 * H2 * 2);
  __bf16* Wm1p   = (__bf16*)carve((size_t)(2 * H2) * MH * 2);
  (void)ws_size; (void)n_in;

  const int TB = 256;

  // --- prep: zero accumulators, pack weights & features, degrees ---
  hipMemsetAsync(deg, 0, (size_t)N * 4, stream);
  hipMemsetAsync(aggf, 0, (size_t)N * IN * 4, stream);

  pack_bf16_kernel<<<(int)cdiv((long)N * IN, TB), TB, 0, stream>>>(inputs, Xbf, (long)N * IN);
  pack_w_kernel<<<(int)cdiv((long)IN * H1, TB), TB, 0, stream>>>(W_self1, Ws1p, IN, H1);
  pack_w_kernel<<<(int)cdiv((long)IN * H1, TB), TB, 0, stream>>>(W_neigh1, Wn1p, IN, H1);
  pack_w_kernel<<<(int)cdiv((long)H1 * H2, TB), TB, 0, stream>>>(W_self2, Ws2p, H1, H2);
  pack_w_kernel<<<(int)cdiv((long)H1 * H2, TB), TB, 0, stream>>>(W_neigh2, Wn2p, H1, H2);
  pack_w_kernel<<<(int)cdiv((long)2 * H2 * MH, TB), TB, 0, stream>>>(W_mlp1, Wm1p, 2 * H2, MH);

  deg_count_kernel<<<(int)cdiv(E1, TB), TB, 0, stream>>>(edge_dst, deg, E1);
  inv_deg_kernel<<<(int)cdiv(N, TB), TB, 0, stream>>>(deg, N);

  // --- layer 1: scatter, scale+pack, WMMA GEMM (persistent, LDS weights) ---
  scatter_f32_kernel<<<(int)cdiv((long)E1 * (IN / 4), TB), TB, 0, stream>>>(
      inputs, edge_src, edge_dst, aggf, E1, IN);
  scale_pack_kernel<<<(int)cdiv((long)N * IN, TB), TB, 0, stream>>>(aggf, deg, AGGbf, N, IN);
  {
    long tiles = cdiv(N, 16);
    int grid = (int)cdiv(tiles, 8);
    if (grid > 640) grid = 640;
    sage_gemm_kernel<8, false><<<grid, 256, 0, stream>>>(
        Xbf, AGGbf, Ws1p, Wn1p, b1, H1bf, nullptr, N);
  }

  // --- layer 2: re-zero accumulators, scatter (bf16 src), scale+pack, GEMM ---
  hipMemsetAsync(aggf, 0, (size_t)N * H1 * 4, stream);
  scatter_bf16_kernel<<<(int)cdiv((long)E1 * (H1 / 4), TB), TB, 0, stream>>>(
      H1bf, edge_src, edge_dst, aggf, E1, H1);
  scale_pack_kernel<<<(int)cdiv((long)N * H1, TB), TB, 0, stream>>>(aggf, deg, AGGbf, N, H1);
  {
    long tiles = cdiv(N, 16);
    int grid = (int)cdiv(tiles, 8);
    if (grid > 640) grid = 640;
    sage_gemm_kernel<4, true><<<grid, 256, 0, stream>>>(
        H1bf, AGGbf, Ws2p, Wn2p, b2, Hbf, out_h, N);
  }

  // --- edge scoring MLP (persistent, LDS weights, WMMA + reduce + sigmoid) ---
  {
    long tiles = cdiv(E2, 16);
    int grid = (int)cdiv(tiles, 8);
    if (grid > 1280) grid = 1280;
    edge_score_kernel<<<grid, 256, 0, stream>>>(
        Hbf, score_src, score_dst, Wm1p, b_mlp1, W_mlp2, b_mlp2, out_score, E2);
  }
  (void)out_size;
}
